// Double_SSM_Block_out_59382217834755
// MI455X (gfx1250) — matrix operand order repaired and checked by reading, not compiled, not run on hardware
//
#include <hip/hip_runtime.h>
#include <hip/hip_bf16.h>
#include <stdint.h>

// ---- Mamba hyperparams (fixed by the reference) ----
#define BATCH   16
#define DMODEL  128
#define HH      64
#define WWID    64
#define LLEN    4096          // HH*WWID
#define MROWS   65536         // BATCH*LLEN
#define DINNER  256
#define DSTATE  16
#define DTRANK  8
#define DCONV   4
#define NXZ     512           // 2*DINNER
#define NPROJ   40            // DTRANK + 2*DSTATE
#define OUTC    128

typedef __attribute__((ext_vector_type(16))) __bf16 v16bf;
typedef __attribute__((ext_vector_type(8)))  float  v8f;
typedef __attribute__((ext_vector_type(4)))  unsigned int v4u;
typedef __attribute__((ext_vector_type(8)))  int    v8i;
typedef __attribute__((ext_vector_type(4)))  int    v4i;

// ---- Tensor Data Mover availability / arity switch ------------------
#if defined(__AMDGCN__) && __has_builtin(__builtin_amdgcn_tensor_load_to_lds) && \
    __has_builtin(__builtin_amdgcn_s_wait_tensorcnt)
#define USE_TDM 1
#else
#define USE_TDM 0
#endif
#if __has_include(<hip/amd_detail/amd_gfx1250_TDM.h>)
#define TDM_SIX_ARG 1
#else
#define TDM_SIX_ARG 0
#endif

#if USE_TDM
// 1D (single-row) tile copy global->LDS via TDM.  D# packed per ISA 8.3/8.4:
//  group0: [1:0]=count=1, [63:32]=lds_addr, [120:64]=global_addr, [127:126]=type=2
//  group1: data_size=1 (2 bytes), tensor_dim0=tile_dim0=nelems, dim1=tile1=1,
//          tensor_dim0_stride=nelems, no padding / multicast / iterate.
__device__ __forceinline__ void tdm_load_row_bf16(uint32_t lds_off,
                                                  const __bf16* gptr,
                                                  uint32_t nelems) {
    uint64_t ga = (uint64_t)(uintptr_t)gptr;
    v4u g0;
    g0[0] = 1u;                                   // count=1 valid descriptor
    g0[1] = lds_off;                              // lds_addr
    g0[2] = (uint32_t)ga;                         // global_addr[31:0]
    g0[3] = (uint32_t)((ga >> 32) & 0x01FFFFFFu)  // global_addr[56:32]
          | (2u << 30);                           // type = 2 ("image")
    v8i g1;
    g1[0] = (int)(1u << 16);                      // data_size=1 (2 bytes/elem)
    g1[1] = (int)((nelems & 0xFFFFu) << 16);      // tensor_dim0 lo16 (bits 63:48)
    g1[2] = (int)((nelems >> 16) | (1u << 16));   // dim0 hi16 | tensor_dim1 lo16=1
    g1[3] = (int)((nelems & 0xFFFFu) << 16);      // dim1 hi=0 | tile_dim0
    g1[4] = 1;                                    // tile_dim1=1 | tile_dim2=0
    g1[5] = (int)nelems;                          // tensor_dim0_stride lo32
    g1[6] = 0;                                    // stride0 hi | stride1 lo
    g1[7] = 0;
    v4i z4 = {};
#if TDM_SIX_ARG
    v8i z8 = {};
    __builtin_amdgcn_tensor_load_to_lds(g0, g1, z4, z4, z8, 0);
#else
    __builtin_amdgcn_tensor_load_to_lds(g0, g1, z4, z4, 0);
#endif
}
#endif

// ------------------------------------------------------------------
// 0a. u = permute(x) cast to bf16:  u[m, c] = x[b, c, l],  m = b*L + l
// ------------------------------------------------------------------
__global__ __launch_bounds__(256) void k_cast_u(const float* __restrict__ x,
                                                __bf16* __restrict__ u_bf) {
    int idx = blockIdx.x * 256 + threadIdx.x;      // MROWS*DMODEL threads
    int m = idx >> 7;
    int c = idx & 127;
    int b = m >> 12;
    int l = m & 4095;
    u_bf[idx] = (__bf16)x[((b * DMODEL + c) << 12) + l];
}

// ------------------------------------------------------------------
// 0b. weights -> bf16, transposed so K is contiguous per output column
// ------------------------------------------------------------------
__global__ __launch_bounds__(256) void k_cast_w(const float* __restrict__ W_in,
                                                const float* __restrict__ W_out,
                                                __bf16* __restrict__ wtin,
                                                __bf16* __restrict__ wtout) {
    int idx = blockIdx.x * 256 + threadIdx.x;
    if (idx < NXZ * DMODEL) {
        int n = idx >> 7, k = idx & 127;
        wtin[idx] = (__bf16)W_in[k * NXZ + n];
    } else if (idx < NXZ * DMODEL + OUTC * DINNER) {
        int j = idx - NXZ * DMODEL;
        int n = j >> 8, k = j & 255;
        wtout[j] = (__bf16)W_out[k * OUTC + n];
    }
}

// ------------------------------------------------------------------
// Shared helper: load one A fragment (16-bit WMMA layout) from LDS.
//  lane 0-15 / 16-31 both hold M=0..15; kh selects the K-half.
//  VGPR v<4 -> K = 2v + 8*kh; v>=4 -> K = 16 + 2(v-4) + 8*kh
// ------------------------------------------------------------------
__device__ __forceinline__ v16bf lds_a_frag(const __bf16* lds, int row, int lda,
                                            int ks, int kh) {
    v16bf a;
#pragma unroll
    for (int e = 0; e < 16; ++e) {
        int v = e >> 1, lo = e & 1;
        int ka = ((v < 4) ? (v << 1) : (16 + ((v - 4) << 1))) + (kh << 3) + lo;
        a[e] = lds[row * lda + ks + ka];
    }
    return a;
}

// ------------------------------------------------------------------
// 1. in_proj GEMM:  xz = u @ W_in   (M=65536, N=512, K=128), bf16 WMMA
//    One block = one 16-row panel (staged once into LDS via TDM) x 32
//    column tiles; 8 waves x 4 tiles each.  A fragments held in regs.
// ------------------------------------------------------------------
__global__ __launch_bounds__(256) void k_gemm_in(const __bf16* __restrict__ u,
                                                 const __bf16* __restrict__ wt,
                                                 __bf16* __restrict__ xs,
                                                 __bf16* __restrict__ zb) {
    __shared__ __bf16 Atile[16 * DMODEL];          // 4 KB panel
    int lane = threadIdx.x & 31;
    int wave = threadIdx.x >> 5;
    int tm   = blockIdx.x;                         // 4096 row tiles

#if USE_TDM
    if (wave == 0) {
        tdm_load_row_bf16((uint32_t)(uintptr_t)&Atile[0],
                          u + (size_t)tm * 16 * DMODEL, 16 * DMODEL);
        __builtin_amdgcn_s_wait_tensorcnt(0);
    }
#else
    for (int i = threadIdx.x; i < 16 * DMODEL; i += 256)
        Atile[i] = u[(size_t)tm * 16 * DMODEL + i];
#endif
    __syncthreads();

    int kh  = lane >> 4;
    int r16 = lane & 15;
    // A fragments for all 4 K-steps, loaded once from LDS
    v16bf af[4];
#pragma unroll
    for (int s = 0; s < 4; ++s)
        af[s] = lds_a_frag(Atile, r16, DMODEL, s * 32, kh);

#pragma unroll
    for (int i = 0; i < 4; ++i) {
        int tn = wave * 4 + i;                     // 32 column tiles
        const __bf16* bp = wt + ((tn << 4) + r16) * DMODEL;
        __builtin_prefetch(bp, 0, 1);
        v8f acc = {};
#pragma unroll
        for (int s = 0; s < 4; ++s) {
            v16bf bq;
#pragma unroll
            for (int e = 0; e < 16; ++e)           // B: e<8 -> K = e + 8*kh; e>=8 -> K = 16 + (e-8) + 8*kh
                bq[e] = bp[s * 32 + (kh << 3) + (e & 7) + ((e >> 3) << 4)];
            acc = __builtin_amdgcn_wmma_f32_16x16x32_bf16(false, af[s], false, bq,
                                                          (short)0, acc, false, false);
        }
        // C/D: lanes 0-15 -> M=r, lanes 16-31 -> M=r+8; N = lane%16
        int mbase = (tm << 4) + (kh << 3);
        int nl    = (tn << 4) + r16;
#pragma unroll
        for (int r = 0; r < 8; ++r) {
            int m = mbase + r;
            if (nl < DINNER) xs[m * DINNER + nl]            = (__bf16)acc[r];
            else             zb[m * DINNER + (nl - DINNER)] = (__bf16)acc[r];
        }
    }
}

// ------------------------------------------------------------------
// 2. causal depthwise conv (4 taps along L) + SiLU
// ------------------------------------------------------------------
__global__ __launch_bounds__(256) void k_conv(const __bf16* __restrict__ xs,
                                              const float* __restrict__ cw,
                                              const float* __restrict__ cb,
                                              float* __restrict__ xc_f,
                                              __bf16* __restrict__ xc_b) {
    int idx = blockIdx.x * 256 + threadIdx.x;   // MROWS*DINNER threads
    int d = idx & 255;
    int m = idx >> 8;                           // m = b*L + l
    int l = m & 4095;
    float a = cb[d];
#pragma unroll
    for (int j = 0; j < DCONV; ++j) {
        int ll = l - (DCONV - 1) + j;
        if (ll >= 0)
            a += cw[d * DCONV + j] * (float)xs[(m - (DCONV - 1) + j) * DINNER + d];
    }
    float s = a / (1.f + __expf(-a));           // SiLU
    xc_f[idx] = s;
    xc_b[idx] = (__bf16)s;
}

// ------------------------------------------------------------------
// 3. x_proj:  dbl = xc @ W_xproj   (N=40: dt[0..8), B[8..24), C[24..40))
// ------------------------------------------------------------------
__global__ __launch_bounds__(256) void k_xproj(const __bf16* __restrict__ xc,
                                               const float* __restrict__ Wx,
                                               float* __restrict__ dbl) {
    int idx = blockIdx.x * 256 + threadIdx.x;   // MROWS*NPROJ threads
    if (idx >= MROWS * NPROJ) return;
    int m = idx / NPROJ, col = idx - m * NPROJ;
    const __bf16* p = xc + m * DINNER;
    float a = 0.f;
    for (int k = 0; k < DINNER; ++k)
        a += (float)p[k] * Wx[k * NPROJ + col];
    dbl[idx] = a;
}

// ------------------------------------------------------------------
// 4. dt_proj + softplus:  delta = softplus(dt @ W_dt + b_dt)   (K=8)
// ------------------------------------------------------------------
__global__ __launch_bounds__(256) void k_delta(const float* __restrict__ dbl,
                                               const float* __restrict__ Wdt,
                                               const float* __restrict__ bdt,
                                               float* __restrict__ delta) {
    int idx = blockIdx.x * 256 + threadIdx.x;   // MROWS*DINNER threads
    int m = idx >> 8, d = idx & 255;
    const float* p = dbl + m * NPROJ;
    float a = bdt[d];
#pragma unroll
    for (int r = 0; r < DTRANK; ++r)
        a += p[r] * Wdt[r * DINNER + d];
    delta[idx] = (a > 20.f) ? a : log1pf(__expf(a));
}

// ------------------------------------------------------------------
// 5. selective scan, fused D-skip + SiLU gate (wave32: 2 channels/wave,
//    lane%16 = state index; shfl_xor reduction within each 16-lane half)
// ------------------------------------------------------------------
__global__ __launch_bounds__(256) void k_scan(const float* __restrict__ delta,
                                              const float* __restrict__ xc_f,
                                              const float* __restrict__ dbl,
                                              const __bf16* __restrict__ zb,
                                              const float* __restrict__ A_log,
                                              const float* __restrict__ Dp,
                                              __bf16* __restrict__ yb) {
    int lane = threadIdx.x & 31;
    int wave = threadIdx.x >> 5;
    int gw = blockIdx.x * 8 + wave;             // 2048 waves total
    int b  = gw >> 7;                           // 128 waves per batch
    int d  = ((gw & 127) << 1) + (lane >> 4);   // 2 channels per wave
    int n  = lane & 15;
    float a  = -__expf(A_log[d * DSTATE + n]);  // A = -exp(A_log)
    float Dd = Dp[d];
    float h = 0.f;
    int mb = b << 12;
    for (int l = 0; l < LLEN; ++l) {
        int m = mb + l;
        float dt = delta[m * DINNER + d];
        float u  = xc_f[m * DINNER + d];
        float Bv = dbl[m * NPROJ + DTRANK + n];
        float Cv = dbl[m * NPROJ + DTRANK + DSTATE + n];
        h = __expf(dt * a) * h + (dt * u) * Bv;
        float p = h * Cv;
        p += __shfl_xor(p, 1, 32);
        p += __shfl_xor(p, 2, 32);
        p += __shfl_xor(p, 4, 32);
        p += __shfl_xor(p, 8, 32);
        if (n == 0) {
            float y  = p + u * Dd;              // D * u skip
            float zv = (float)zb[m * DINNER + d];
            y *= zv / (1.f + __expf(-zv));      // y * silu(z)
            yb[m * DINNER + d] = (__bf16)y;
        }
    }
}

// ------------------------------------------------------------------
// 6. out_proj GEMM:  out = y @ W_out  (M=65536, N=128, K=256), bf16 WMMA
//    One block = one 16-row panel (TDM -> LDS, 8 KB) x 8 column tiles;
//    final (B, OC, W, H) permute fused into the accumulator store.
// ------------------------------------------------------------------
__global__ __launch_bounds__(256) void k_gemm_out(const __bf16* __restrict__ y,
                                                  const __bf16* __restrict__ wt,
                                                  float* __restrict__ out) {
    __shared__ __bf16 Atile[16 * DINNER];          // 8 KB panel
    int lane = threadIdx.x & 31;
    int wave = threadIdx.x >> 5;
    int tm   = blockIdx.x;                         // 4096 row tiles

#if USE_TDM
    if (wave == 0) {
        tdm_load_row_bf16((uint32_t)(uintptr_t)&Atile[0],
                          y + (size_t)tm * 16 * DINNER, 16 * DINNER);
        __builtin_amdgcn_s_wait_tensorcnt(0);
    }
#else
    for (int i = threadIdx.x; i < 16 * DINNER; i += 256)
        Atile[i] = y[(size_t)tm * 16 * DINNER + i];
#endif
    __syncthreads();

    int kh  = lane >> 4;
    int r16 = lane & 15;
    v16bf af[8];
#pragma unroll
    for (int s = 0; s < 8; ++s)
        af[s] = lds_a_frag(Atile, r16, DINNER, s * 32, kh);

    int tn = wave;                                 // 8 column tiles
    const __bf16* bp = wt + ((tn << 4) + r16) * DINNER;
    __builtin_prefetch(bp, 0, 1);
    v8f acc = {};
#pragma unroll
    for (int s = 0; s < 8; ++s) {
        v16bf bq;
#pragma unroll
        for (int e = 0; e < 16; ++e)
            bq[e] = bp[s * 32 + (kh << 3) + (e & 7) + ((e >> 3) << 4)];
        acc = __builtin_amdgcn_wmma_f32_16x16x32_bf16(false, af[s], false, bq,
                                                      (short)0, acc, false, false);
    }
    int mbase = (tm << 4) + (kh << 3);
    int oc    = (tn << 4) + r16;
#pragma unroll
    for (int r = 0; r < 8; ++r) {
        int m = mbase + r;
        int b = m >> 12, l = m & 4095;
        int hh = l >> 6, w2 = l & 63;
        out[((b * OUTC + oc) * WWID + w2) * HH + hh] = acc[r];   // out[b,oc,w,h]
    }
}

// ------------------------------------------------------------------
extern "C" void kernel_launch(void* const* d_in, const int* in_sizes, int n_in,
                              void* d_out, int out_size, void* d_ws, size_t ws_size,
                              hipStream_t stream) {
    const float* x      = (const float*)d_in[0];
    const float* W_in   = (const float*)d_in[1];
    const float* conv_w = (const float*)d_in[2];
    const float* conv_b = (const float*)d_in[3];
    const float* W_xp   = (const float*)d_in[4];
    const float* W_dt   = (const float*)d_in[5];
    const float* b_dt   = (const float*)d_in[6];
    const float* A_log  = (const float*)d_in[7];
    const float* D_par  = (const float*)d_in[8];
    const float* W_out  = (const float*)d_in[9];
    float* out = (float*)d_out;

    char* ws = (char*)d_ws;
    size_t off = 0;
    auto alloc = [&](size_t bytes) -> void* {
        off = (off + 255) & ~(size_t)255;
        void* p = ws + off;
        off += bytes;
        return p;
    };
    __bf16* u_bf  = (__bf16*)alloc((size_t)MROWS * DMODEL * 2);   // 16 MB
    __bf16* wtin  = (__bf16*)alloc((size_t)NXZ * DMODEL * 2);
    __bf16* wtout = (__bf16*)alloc((size_t)OUTC * DINNER * 2);
    __bf16* xs_bf = (__bf16*)alloc((size_t)MROWS * DINNER * 2);   // 32 MB
    __bf16* z_bf  = (__bf16*)alloc((size_t)MROWS * DINNER * 2);   // 32 MB
    float*  xc_f  = (float*) alloc((size_t)MROWS * DINNER * 4);   // 64 MB
    __bf16* xc_bf = (__bf16*)alloc((size_t)MROWS * DINNER * 2);   // 32 MB
    float*  dbl   = (float*) alloc((size_t)MROWS * NPROJ  * 4);   // 10 MB
    float*  delta = (float*) alloc((size_t)MROWS * DINNER * 4);   // 64 MB
    __bf16* y_bf  = (__bf16*)alloc((size_t)MROWS * DINNER * 2);   // 32 MB

    k_cast_u<<<(MROWS * DMODEL) / 256, 256, 0, stream>>>(x, u_bf);
    k_cast_w<<<(NXZ * DMODEL + OUTC * DINNER + 255) / 256, 256, 0, stream>>>(
        W_in, W_out, wtin, wtout);
    // in_proj: one block per 16-row panel (4096 blocks)
    k_gemm_in<<<MROWS / 16, 256, 0, stream>>>(u_bf, wtin, xs_bf, z_bf);
    k_conv<<<(MROWS * DINNER) / 256, 256, 0, stream>>>(xs_bf, conv_w, conv_b, xc_f, xc_bf);
    k_xproj<<<(MROWS * NPROJ + 255) / 256, 256, 0, stream>>>(xc_bf, W_xp, dbl);
    k_delta<<<(MROWS * DINNER) / 256, 256, 0, stream>>>(dbl, W_dt, b_dt, delta);
    k_scan<<<(BATCH * DINNER / 2) / 8, 256, 0, stream>>>(delta, xc_f, dbl, z_bf,
                                                         A_log, D_par, y_bf);
    // out_proj: one block per 16-row panel (4096 blocks)
    k_gemm_out<<<MROWS / 16, 256, 0, stream>>>(y_bf, wtout, out);
}
